// StripedMamba_2078764171586
// MI455X (gfx1250) — matrix operand order repaired and checked by reading, not compiled, not run on hardware
//
#include <hip/hip_runtime.h>
#include <hip/hip_bf16.h>
#include <math.h>

// ---------------------------------------------------------------------------
// StripedMamba (2-layer Mamba + Enformer-attention hybrid) for gfx1250.
// Dense GEMMs: v_wmma_f32_16x16x32_bf16, LDS-staged (B transposed) with
// ds_load_b128 fragment reads and global_prefetch of the next K tile.
// Attention: flash-style, block-cooperative K/V LDS staging, WMMA for q.k and
// p.V; relative-position basis collapsed to suffix-sum lookups.
// ---------------------------------------------------------------------------

typedef __attribute__((ext_vector_type(16))) __bf16 v16bf;
typedef __attribute__((ext_vector_type(8)))  __bf16 v8bf;
typedef __attribute__((ext_vector_type(8)))  float  v8f;

#define SEQB 1024
#define DMODEL 512
#define MROWS 2048   // B * SEQ

static __device__ inline v8f wmma_bf16(v16bf a, v16bf b, v8f c) {
  return __builtin_amdgcn_wmma_f32_16x16x32_bf16(false, a, false, b, (short)0, c,
                                                 false, false);
}

static __device__ inline v16bf concat8(v8bf a, v8bf b) {
  return __builtin_shufflevector(a, b, 0, 1, 2, 3, 4, 5, 6, 7,
                                 8, 9, 10, 11, 12, 13, 14, 15);
}

// load 8 consecutive f32 (16B-aligned) and convert to bf16x8
static __device__ inline v8bf cvt8(const float* __restrict__ p) {
  float4 x = *(const float4*)p;
  float4 y = *(const float4*)(p + 4);
  v8bf r;
  r[0] = (__bf16)x.x; r[1] = (__bf16)x.y; r[2] = (__bf16)x.z; r[3] = (__bf16)x.w;
  r[4] = (__bf16)y.x; r[5] = (__bf16)y.y; r[6] = (__bf16)y.z; r[7] = (__bf16)y.w;
  return r;
}

static __device__ inline float silu_f(float x) {
  return x * (1.0f / (1.0f + __expf(-x)));
}

// ======================== generic bf16-WMMA GEMM ===========================
// C[M,N] = A[M,K] @ Bw[K,N] (+bias) (+softplus) (+=C).  M%64==0, N%64==0,
// K%32==0.  Block = 256 threads = 8 waves; block tile 64x64; wave tile 16x32.
__global__ __launch_bounds__(256) void gemm_bf16_wmma(
    const float* __restrict__ A, int lda,
    const float* __restrict__ Bw, int ldb,
    float* __restrict__ C, int ldc,
    int K, const float* __restrict__ bias_p, int epilogue, int addC) {
  __shared__ __bf16 sA[64][40];    // [m][k], 80B row stride (16B aligned)
  __shared__ __bf16 sBt[64][40];   // [n][k], transposed for vector frag reads

  const int tileN = blockIdx.x * 64;
  const int tileM = blockIdx.y * 64;
  const int tid = threadIdx.x;
  const int lane = tid & 31, w = tid >> 5;
  const int wm = w >> 1, wn = w & 1;        // 4x2 wave grid, wave tile 16x32
  const int g = lane >> 4, r = lane & 15;

  const int arow = tid >> 2, ak = (tid & 3) * 8;   // A: 64 rows x 32 k, 8/thread
  const int brow = tid >> 3, bn = (tid & 7) * 8;   // B: 32 k x 64 n, 8/thread

  v8f acc0 = {}, acc1 = {};

  for (int k0 = 0; k0 < K; k0 += 32) {
    const float* ap = A + (size_t)(tileM + arow) * lda + k0 + ak;
    float4 a0 = *(const float4*)ap;
    float4 a1 = *(const float4*)(ap + 4);
    sA[arow][ak + 0] = (__bf16)a0.x;  sA[arow][ak + 1] = (__bf16)a0.y;
    sA[arow][ak + 2] = (__bf16)a0.z;  sA[arow][ak + 3] = (__bf16)a0.w;
    sA[arow][ak + 4] = (__bf16)a1.x;  sA[arow][ak + 5] = (__bf16)a1.y;
    sA[arow][ak + 6] = (__bf16)a1.z;  sA[arow][ak + 7] = (__bf16)a1.w;
    const float* bp = Bw + (size_t)(k0 + brow) * ldb + tileN + bn;
    float4 b0 = *(const float4*)bp;
    float4 b1 = *(const float4*)(bp + 4);
    sBt[bn + 0][brow] = (__bf16)b0.x;  sBt[bn + 1][brow] = (__bf16)b0.y;
    sBt[bn + 2][brow] = (__bf16)b0.z;  sBt[bn + 3][brow] = (__bf16)b0.w;
    sBt[bn + 4][brow] = (__bf16)b1.x;  sBt[bn + 5][brow] = (__bf16)b1.y;
    sBt[bn + 6][brow] = (__bf16)b1.z;  sBt[bn + 7][brow] = (__bf16)b1.w;
    __syncthreads();

    if (k0 + 32 < K) {                       // gfx1250 global_prefetch_b8
      __builtin_prefetch(ap + 32, 0, 1);
      __builtin_prefetch(bp + (size_t)32 * ldb, 0, 1);
    }

    // fragments via 16B ds_load_b128 reads
    const v8bf* pa = (const v8bf*)&sA[wm * 16 + r][0];
    v16bf af = concat8(pa[2 * g], pa[2 * g + 1]);       // k = 16g..16g+15
    const v8bf* pb0 = (const v8bf*)&sBt[wn * 32 + r][0];
    const v8bf* pb1 = (const v8bf*)&sBt[wn * 32 + 16 + r][0];
    v16bf bf0 = concat8(pb0[2 * g], pb0[2 * g + 1]);    // k = 16g..16g+15
    v16bf bf1 = concat8(pb1[2 * g], pb1[2 * g + 1]);
    acc0 = wmma_bf16(af, bf0, acc0);
    acc1 = wmma_bf16(af, bf1, acc1);
    __syncthreads();
  }

#pragma unroll
  for (int vi = 0; vi < 8; ++vi) {
    int row = tileM + wm * 16 + vi + 8 * g;   // C layout: row = vi + 8g
    int col0 = tileN + wn * 32 + r;
    float v0 = acc0[vi], v1 = acc1[vi];
    if (bias_p) { v0 += bias_p[col0]; v1 += bias_p[col0 + 16]; }
    if (epilogue == 1) {   // softplus (fast path)
      v0 = (v0 > 20.f) ? v0 : __logf(1.f + __expf(v0));
      v1 = (v1 > 20.f) ? v1 : __logf(1.f + __expf(v1));
    }
    float* cp = C + (size_t)row * ldc + col0;
    if (addC) { v0 += cp[0]; v1 += cp[16]; }
    cp[0] = v0;
    cp[16] = v1;
  }
}

// ============================ small kernels ================================
__global__ __launch_bounds__(256) void rms_kernel(const float* __restrict__ X,
                                                  const float* __restrict__ Wt,
                                                  float* __restrict__ Y) {
  int row = blockIdx.x, tid = threadIdx.x;
  float a = X[(size_t)row * DMODEL + tid];
  float b = X[(size_t)row * DMODEL + 256 + tid];
  __shared__ float red[256];
  red[tid] = a * a + b * b;
  __syncthreads();
  for (int s = 128; s > 0; s >>= 1) {
    if (tid < s) red[tid] += red[tid + s];
    __syncthreads();
  }
  float rr = rsqrtf(red[0] / (float)DMODEL + 1e-6f);
  Y[(size_t)row * DMODEL + tid] = a * rr * Wt[tid];
  Y[(size_t)row * DMODEL + 256 + tid] = b * rr * Wt[256 + tid];
}

// causal depthwise conv (KC=4) over xc half of XZ, + bias, + SiLU
__global__ __launch_bounds__(256) void conv_silu_kernel(
    const float* __restrict__ XZ, const float* __restrict__ cw,
    const float* __restrict__ cb, float* __restrict__ XC) {
  int id = blockIdx.x * 256 + threadIdx.x;     // MROWS*DMODEL
  int row = id >> 9, d = id & 511;
  int b = row >> 10, t = row & 1023;
  float acc = cb[d];
#pragma unroll
  for (int k = 0; k < 4; ++k) {
    int ts = t - 3 + k;
    if (ts >= 0) acc += XZ[(size_t)(b * SEQB + ts) * 1024 + d] * cw[d * 4 + k];
  }
  XC[id] = silu_f(acc);
}

// selective scan: one thread per (b,d); 16-state recurrence over 1024 steps.
__global__ __launch_bounds__(256) void scan_kernel(
    const float* __restrict__ XC, const float* __restrict__ XZ,
    const float* __restrict__ DBL, const float* __restrict__ DT,
    const float* __restrict__ Alog, const float* __restrict__ Dp,
    float* __restrict__ YS) {
  int id = blockIdx.x * 256 + threadIdx.x;
  if (id >= 1024) return;
  int b = id >> 9, d = id & 511;
  float A[16], hs[16];
#pragma unroll
  for (int n = 0; n < 16; ++n) {
    A[n] = -__expf(Alog[d * 16 + n]);
    hs[n] = 0.f;
  }
  float Dd = Dp[d];
  for (int t = 0; t < SEQB; ++t) {
    int row = b * SEQB + t;
    float dtv = DT[(size_t)row * DMODEL + d];
    float xcv = XC[(size_t)row * DMODEL + d];
    const float* bl = DBL + (size_t)row * 64;
    float y = 0.f;
#pragma unroll
    for (int n = 0; n < 16; ++n) {
      hs[n] = __expf(dtv * A[n]) * hs[n] + dtv * bl[32 + n] * xcv;
      y += hs[n] * bl[48 + n];
    }
    float z = XZ[(size_t)row * 1024 + 512 + d];
    YS[(size_t)row * DMODEL + d] = (y + Dd * xcv) * silu_f(z);
  }
}

__global__ __launch_bounds__(256) void flip_kernel(const float* __restrict__ In,
                                                   int lda, int coff,
                                                   float* __restrict__ Out) {
  int id = blockIdx.x * 256 + threadIdx.x;
  int row = id >> 9, d = id & 511;
  int b = row >> 10, t = row & 1023;
  Out[id] = In[(size_t)(b * SEQB + (1023 - t)) * lda + coff + d];
}

__global__ __launch_bounds__(256) void combine_kernel(
    const float* __restrict__ YF, const float* __restrict__ YR,
    const float* __restrict__ PROJ, float* __restrict__ YC) {
  int id = blockIdx.x * 256 + threadIdx.x;
  int row = id >> 9, d = id & 511;
  float zf = PROJ[(size_t)row * 1536 + 512 + d];
  float zr = PROJ[(size_t)row * 1536 + 1024 + d];
  YC[id] = YF[id] * silu_f(zf) + YR[id] * silu_f(zr);
}

// uk[b,h,j] = u[h,:] . k[b,j,h,:]
__global__ __launch_bounds__(256) void uk_kernel(const float* __restrict__ U,
                                                 const float* __restrict__ Kb,
                                                 float* __restrict__ UK) {
  int id = blockIdx.x * 256 + threadIdx.x;   // B*H*SEQ = 16384
  int b = id >> 13, h = (id >> 10) & 7, j = id & 1023;
  const float* kr = Kb + (size_t)(b * SEQB + j) * DMODEL + h * 64;
  const float* ur = U + h * 64;
  float s = 0.f;
#pragma unroll 8
  for (int dd = 0; dd < 64; ++dd) s += ur[dd] * kr[dd];
  UK[id] = s;
}

// qw[b,h,i,:] = q . w_rel  then suffix sums (2 halves x 17) -> SUFQ
__global__ __launch_bounds__(256) void sufq_kernel(const float* __restrict__ Q,
                                                   const float* __restrict__ Wrel,
                                                   float* __restrict__ SUFQ) {
  int id = blockIdx.x * 256 + threadIdx.x;   // 16384 = bh*1024 + i
  int b = id >> 13, h = (id >> 10) & 7, i = id & 1023;
  const float* qrow = Q + (size_t)(b * SEQB + i) * DMODEL + h * 64;
  const float* wr = Wrel + (size_t)h * 64 * 32;
  float qw[32];
#pragma unroll
  for (int n = 0; n < 32; ++n) qw[n] = 0.f;
  for (int dd = 0; dd < 64; ++dd) {
    float qv = qrow[dd];
    const float* wd = wr + dd * 32;
#pragma unroll
    for (int n = 0; n < 32; ++n) qw[n] += qv * wd[n];
  }
  float* out = SUFQ + (size_t)id * 34;
  out[16] = 0.f;
  out[33] = 0.f;
  float su = 0.f, ss = 0.f;
  for (int t = 15; t >= 0; --t) {
    su += qw[t];
    ss += qw[16 + t];
    out[t] = su;
    out[17 + t] = ss;
  }
}

// vn[h,n] = vb[h,:].w_rel[h,:,n]; suffix sums -> SUFV[h][2][17]
__global__ __launch_bounds__(256) void vnsuf_kernel(const float* __restrict__ Vbias,
                                                    const float* __restrict__ Wrel,
                                                    float* __restrict__ SUFV) {
  __shared__ float vn[256];
  int tid = threadIdx.x;
  int h = tid >> 5, n = tid & 31;
  float s = 0.f;
  for (int dd = 0; dd < 64; ++dd)
    s += Vbias[h * 64 + dd] * Wrel[(size_t)(h * 64 + dd) * 32 + n];
  vn[tid] = s;
  __syncthreads();
  if (tid < 8) {
    float* out = SUFV + tid * 34;
    out[16] = 0.f;
    out[33] = 0.f;
    float su = 0.f, ss = 0.f;
    for (int t = 15; t >= 0; --t) {
      su += vn[tid * 32 + t];
      ss += vn[tid * 32 + 16 + t];
      out[t] = su;
      out[17 + t] = ss;
    }
  }
}

// tTab[a] = smallest n with a <= cw[n] (16 if none); cw[n] = pr^(n+1)
__global__ __launch_bounds__(256) void ttab_kernel(int* __restrict__ T) {
  int id = blockIdx.x * 256 + threadIdx.x;
  if (id >= 1024) return;
  float lpr = logf((SEQB + 1) / 2.0f) / 16.0f;   // log(pow_rate)
  float a = (float)id;
  int t = 16;
  for (int n = 15; n >= 0; --n) {
    float cw = __expf((float)(n + 1) * lpr);
    if (a <= cw) t = n;
  }
  T[id] = t;
}

// ===================== flash attention with WMMA ===========================
// Block = 8 waves; each wave owns a 16-row i-tile (block covers 128 rows of
// one (b,h)).  K/V tiles (32 j-rows) staged cooperatively in LDS per chunk
// (V transposed); SUFQ / tTab / v-bias suffix tables staged in LDS up front.
__global__ __launch_bounds__(256) void attn_kernel(
    const float* __restrict__ Q, const float* __restrict__ Kb,
    const float* __restrict__ Vb, const float* __restrict__ UK,
    const float* __restrict__ SUFQ, const float* __restrict__ SUFV,
    const int* __restrict__ TTAB, float* __restrict__ O) {
  int bid = blockIdx.x;
  int bh = bid >> 3, chunk = bid & 7;
  int b = bh >> 3, h = bh & 7;
  int tid = threadIdx.x, lane = tid & 31, w = tid >> 5;
  int g = lane >> 4, r = lane & 15;
  int itile = chunk * 128 + w * 16;

  __shared__ __bf16 sK[32][72];        // [j][dh], 144B rows (16B aligned)
  __shared__ __bf16 sVt[64][40];       // [dh][j], transposed
  __shared__ __bf16 pbuf[8][16][32];   // per-wave p transpose buffer
  __shared__ float  sufq[8 * 544];     // per-wave 16 rows x 34 suffix entries
  __shared__ float  sVb[34];
  __shared__ int    sTT[1024];

  for (int idx = tid; idx < 1024; idx += 256) sTT[idx] = TTAB[idx];
  if (tid < 34) sVb[tid] = SUFV[h * 34 + tid];
  for (int idx = lane; idx < 544; idx += 32)
    sufq[w * 544 + idx] = SUFQ[(size_t)(bh * SEQB + itile) * 34 + idx];
  __syncthreads();

  const float scale = 0.125f;   // 1/sqrt(64)

  // q A-fragments (dh chunks [0,32) and [32,64)) via float4 global loads
  const float* qbase = Q + (size_t)(b * SEQB + itile + r) * DMODEL + h * 64;
  v16bf qa0 = concat8(cvt8(qbase + 8 * g), cvt8(qbase + 16 + 8 * g));
  v16bf qa1 = concat8(cvt8(qbase + 32 + 8 * g), cvt8(qbase + 48 + 8 * g));

  v8f oacc[4];
#pragma unroll
  for (int nt = 0; nt < 4; ++nt) { v8f z = {}; oacc[nt] = z; }
  float mrow[8], srow[8];
#pragma unroll
  for (int vi = 0; vi < 8; ++vi) { mrow[vi] = -3.0e38f; srow[vi] = 0.f; }

  const int krow = tid >> 3, kc8 = (tid & 7) * 8;   // staging: 32 rows x 8 dh

  for (int jc = 0; jc < SEQB; jc += 32) {
    // ---- cooperative K/V staging (f32 -> bf16) ----
    {
      const float* kp = Kb + (size_t)(b * SEQB + jc + krow) * DMODEL + h * 64 + kc8;
      float4 k0 = *(const float4*)kp;
      float4 k1 = *(const float4*)(kp + 4);
      sK[krow][kc8 + 0] = (__bf16)k0.x;  sK[krow][kc8 + 1] = (__bf16)k0.y;
      sK[krow][kc8 + 2] = (__bf16)k0.z;  sK[krow][kc8 + 3] = (__bf16)k0.w;
      sK[krow][kc8 + 4] = (__bf16)k1.x;  sK[krow][kc8 + 5] = (__bf16)k1.y;
      sK[krow][kc8 + 6] = (__bf16)k1.z;  sK[krow][kc8 + 7] = (__bf16)k1.w;
      const float* vp = Vb + (size_t)(b * SEQB + jc + krow) * DMODEL + h * 64 + kc8;
      float4 v0 = *(const float4*)vp;
      float4 v1 = *(const float4*)(vp + 4);
      sVt[kc8 + 0][krow] = (__bf16)v0.x;  sVt[kc8 + 1][krow] = (__bf16)v0.y;
      sVt[kc8 + 2][krow] = (__bf16)v0.z;  sVt[kc8 + 3][krow] = (__bf16)v0.w;
      sVt[kc8 + 4][krow] = (__bf16)v1.x;  sVt[kc8 + 5][krow] = (__bf16)v1.y;
      sVt[kc8 + 6][krow] = (__bf16)v1.z;  sVt[kc8 + 7][krow] = (__bf16)v1.w;
    }
    __syncthreads();

    float ps[2][8];
#pragma unroll
    for (int sj = 0; sj < 2; ++sj) {
      // K B-fragments from LDS (b128 reads)
      const v8bf* pk = (const v8bf*)&sK[sj * 16 + r][0];
      v16bf kf0 = concat8(pk[g], pk[2 + g]);        // dh 0..31
      v16bf kf1 = concat8(pk[4 + g], pk[6 + g]);    // dh 32..63
      v8f s = {};
      s = wmma_bf16(qa0, kf0, s);
      s = wmma_bf16(qa1, kf1, s);
      int j = jc + sj * 16 + r;
      float ukv = UK[bh * SEQB + j];
#pragma unroll
      for (int vi = 0; vi < 8; ++vi) {
        int li = vi + 8 * g;
        int i = itile + li;
        int dd = i - j;
        int a = dd < 0 ? -dd : dd;
        int t = sTT[a];
        float sgn = (dd > 0) ? 1.f : ((dd < 0) ? -1.f : 0.f);
        const float* sq = &sufq[w * 544 + li * 34];
        float biasv = sq[t] + sgn * sq[17 + t] + sVb[t] + sgn * sVb[17 + t] + ukv;
        ps[sj][vi] = (s[vi] + biasv) * scale;
      }
    }
    // ---- online softmax update ----
    float corr[8];
#pragma unroll
    for (int vi = 0; vi < 8; ++vi) {
      float mx = fmaxf(ps[0][vi], ps[1][vi]);
      mx = fmaxf(mx, __shfl_xor(mx, 1, 32));
      mx = fmaxf(mx, __shfl_xor(mx, 2, 32));
      mx = fmaxf(mx, __shfl_xor(mx, 4, 32));
      mx = fmaxf(mx, __shfl_xor(mx, 8, 32));
      float nm = fmaxf(mrow[vi], mx);
      float c = __expf(mrow[vi] - nm);
      float p0 = __expf(ps[0][vi] - nm);
      float p1 = __expf(ps[1][vi] - nm);
      float rs = p0 + p1;
      rs += __shfl_xor(rs, 1, 32);
      rs += __shfl_xor(rs, 2, 32);
      rs += __shfl_xor(rs, 4, 32);
      rs += __shfl_xor(rs, 8, 32);
      srow[vi] = srow[vi] * c + rs;
      mrow[vi] = nm;
      corr[vi] = c;
      pbuf[w][vi + 8 * g][r] = (__bf16)p0;       // C-layout -> LDS (row, jloc)
      pbuf[w][vi + 8 * g][16 + r] = (__bf16)p1;
    }
#pragma unroll
    for (int nt = 0; nt < 4; ++nt)
#pragma unroll
      for (int vi = 0; vi < 8; ++vi) oacc[nt][vi] *= corr[vi];

    // p back as A-fragment (same wave; LDS ops are in-order)
    const v8bf* pp = (const v8bf*)&pbuf[w][r][0];
    v16bf pa = concat8(pp[g], pp[2 + g]);

    // accumulate p @ V over the 32-j chunk, 4 dh tiles of 16
#pragma unroll
    for (int nt = 0; nt < 4; ++nt) {
      const v8bf* pv = (const v8bf*)&sVt[nt * 16 + r][0];
      v16bf vf = concat8(pv[g], pv[2 + g]);
      oacc[nt] = wmma_bf16(pa, vf, oacc[nt]);
    }
    __syncthreads();   // before next chunk overwrites sK/sVt
  }

#pragma unroll
  for (int vi = 0; vi < 8; ++vi) {
    float inv = 1.f / srow[vi];
    size_t base = (size_t)(b * SEQB + itile + vi + 8 * g) * DMODEL + h * 64;
#pragma unroll
    for (int nt = 0; nt < 4; ++nt) O[base + nt * 16 + r] = oacc[nt][vi] * inv;
  }
}

// =============================== host side =================================
static inline void launch_gemm(const float* A, int lda, const float* Bw, int ldb,
                               float* C, int ldc, int M, int N, int K,
                               const float* bias, int epi, int addC,
                               hipStream_t s) {
  dim3 grid(N / 64, M / 64);
  gemm_bf16_wmma<<<grid, 256, 0, s>>>(A, lda, Bw, ldb, C, ldc, K, bias, epi, addC);
}

extern "C" void kernel_launch(void* const* d_in, const int* in_sizes, int n_in,
                              void* d_out, int out_size, void* d_ws, size_t ws_size,
                              hipStream_t stream) {
  (void)in_sizes; (void)n_in; (void)out_size; (void)ws_size;
  const float* in_x    = (const float*)d_in[0];
  const float* in_rmsm = (const float*)d_in[1];
  const float* in_Win  = (const float*)d_in[2];
  const float* in_bin  = (const float*)d_in[3];
  const float* in_mWin = (const float*)d_in[4];
  const float* in_mcw  = (const float*)d_in[5];
  const float* in_mcb  = (const float*)d_in[6];
  const float* in_mWx  = (const float*)d_in[7];
  const float* in_mWdt = (const float*)d_in[8];
  const float* in_mbdt = (const float*)d_in[9];
  const float* in_mAlog= (const float*)d_in[10];
  const float* in_mD   = (const float*)d_in[11];
  const float* in_mWout= (const float*)d_in[12];
  const float* in_Wout = (const float*)d_in[13];
  const float* in_bout = (const float*)d_in[14];
  const float* in_rmsa = (const float*)d_in[15];
  const float* in_Wq   = (const float*)d_in[16];
  const float* in_Wk   = (const float*)d_in[17];
  const float* in_Wv   = (const float*)d_in[18];
  const float* in_u    = (const float*)d_in[19];
  const float* in_v    = (const float*)d_in[20];
  const float* in_wrel = (const float*)d_in[21];
  const float* in_Wo   = (const float*)d_in[22];
  const float* in_bo   = (const float*)d_in[23];

  // workspace arena (floats); ~56.5 MB total. Attention aliases mamba scratch.
  float* ws = (float*)d_ws;
  size_t o = 0;
  float* X    = ws + o; o += (size_t)MROWS * 512;
  float* XN   = ws + o; o += (size_t)MROWS * 512;
  float* PROJ = ws + o; o += (size_t)MROWS * 1536;
  float* XFD  = ws + o; o += (size_t)MROWS * 512;    // also YFD, also Q
  float* XZ   = ws + o; o += (size_t)MROWS * 1024;   // also K
  float* XC   = ws + o; o += (size_t)MROWS * 512;    // also V
  float* DBL  = ws + o; o += (size_t)MROWS * 64;
  float* DT   = ws + o; o += (size_t)MROWS * 512;    // also SUFQ (557056 fits)
  float* YS   = ws + o; o += (size_t)MROWS * 512;    // also attention O
  float* YF   = ws + o; o += (size_t)MROWS * 512;    // also UK (16384 fits)
  float* YR   = ws + o; o += (size_t)MROWS * 512;
  float* YC   = ws + o; o += (size_t)MROWS * 512;
  float* SUFV = ws + o; o += 8 * 34;
  int*   TTAB = (int*)(ws + o); o += 1024;

  float* Qb   = XFD;
  float* Kbuf = XZ;
  float* Vbuf = XC;
  float* Ob   = YS;
  float* UK   = YF;
  float* SUFQ = DT;

  const size_t xbytes = (size_t)MROWS * 512 * sizeof(float);
  hipMemcpyAsync(X, in_x, xbytes, hipMemcpyDeviceToDevice, stream);
  ttab_kernel<<<4, 256, 0, stream>>>(TTAB);

  const int NEL = MROWS * 512;            // 1M elementwise grid
  const int EB  = NEL / 256;              // 4096 blocks

  for (int L = 0; L < 2; ++L) {
    // ---------------- Mamba block ----------------
    rms_kernel<<<MROWS, 256, 0, stream>>>(X, in_rmsm + L * 512, XN);
    launch_gemm(XN, 512, in_Win + (size_t)L * 512 * 1536, 1536, PROJ, 1536,
                MROWS, 1536, 512, in_bin + L * 1536, 0, 0, stream);

    for (int dir = 0; dir < 2; ++dir) {
      int s2 = L * 2 + dir;
      const float* Ain; int lda;
      if (dir == 0) { Ain = PROJ; lda = 1536; }
      else {
        flip_kernel<<<EB, 256, 0, stream>>>(PROJ, 1536, 0, XFD);
        Ain = XFD; lda = 512;
      }
      launch_gemm(Ain, lda, in_mWin + (size_t)s2 * 512 * 1024, 1024, XZ, 1024,
                  MROWS, 1024, 512, nullptr, 0, 0, stream);
      conv_silu_kernel<<<EB, 256, 0, stream>>>(XZ, in_mcw + (size_t)s2 * 512 * 4,
                                               in_mcb + (size_t)s2 * 512, XC);
      launch_gemm(XC, 512, in_mWx + (size_t)s2 * 512 * 64, 64, DBL, 64,
                  MROWS, 64, 512, nullptr, 0, 0, stream);
      launch_gemm(DBL, 64, in_mWdt + (size_t)s2 * 32 * 512, 512, DT, 512,
                  MROWS, 512, 32, in_mbdt + (size_t)s2 * 512, 1 /*softplus*/, 0,
                  stream);
      scan_kernel<<<4, 256, 0, stream>>>(XC, XZ, DBL, DT,
                                         in_mAlog + (size_t)s2 * 512 * 16,
                                         in_mD + (size_t)s2 * 512, YS);
      if (dir == 0) {
        launch_gemm(YS, 512, in_mWout + (size_t)s2 * 512 * 512, 512, YF, 512,
                    MROWS, 512, 512, nullptr, 0, 0, stream);
      } else {
        launch_gemm(YS, 512, in_mWout + (size_t)s2 * 512 * 512, 512, XFD, 512,
                    MROWS, 512, 512, nullptr, 0, 0, stream);
        flip_kernel<<<EB, 256, 0, stream>>>(XFD, 512, 0, YR);
      }
    }
    combine_kernel<<<EB, 256, 0, stream>>>(YF, YR, PROJ, YC);
    launch_gemm(YC, 512, in_Wout + (size_t)L * 512 * 512, 512, X, 512,
                MROWS, 512, 512, in_bout + L * 512, 0, 1 /*+skip*/, stream);

    // ---------------- Enformer attention ----------------
    rms_kernel<<<MROWS, 256, 0, stream>>>(X, in_rmsa + L * 512, XN);
    launch_gemm(XN, 512, in_Wq + (size_t)L * 512 * 512, 512, Qb, 512,
                MROWS, 512, 512, nullptr, 0, 0, stream);
    launch_gemm(XN, 512, in_Wk + (size_t)L * 512 * 512, 512, Kbuf, 512,
                MROWS, 512, 512, nullptr, 0, 0, stream);
    launch_gemm(XN, 512, in_Wv + (size_t)L * 512 * 512, 512, Vbuf, 512,
                MROWS, 512, 512, nullptr, 0, 0, stream);

    uk_kernel<<<64, 256, 0, stream>>>(in_u + (size_t)L * 8 * 64, Kbuf, UK);
    sufq_kernel<<<64, 256, 0, stream>>>(Qb, in_wrel + (size_t)L * 8 * 64 * 32,
                                        SUFQ);
    vnsuf_kernel<<<1, 256, 0, stream>>>(in_v + (size_t)L * 8 * 64,
                                        in_wrel + (size_t)L * 8 * 64 * 32, SUFV);
    attn_kernel<<<128, 256, 0, stream>>>(Qb, Kbuf, Vbuf, UK, SUFQ, SUFV, TTAB,
                                         Ob);
    launch_gemm(Ob, 512, in_Wo + (size_t)L * 512 * 512, 512, X, 512,
                MROWS, 512, 512, in_bo + L * 512, 0, 1 /*+skip*/, stream);
  }

  hipMemcpyAsync(d_out, X, xbytes, hipMemcpyDeviceToDevice, stream);
}
